// GRU_19696720019463
// MI455X (gfx1250) — matrix operand order repaired and checked by reading, not compiled, not run on hardware
//
#include <hip/hip_runtime.h>
#include <hip/hip_bf16.h>
#include <math.h>

// ---------------------------------------------------------------------------
// GRU cell for MI455X (gfx1250, wave32, WMMA), v2.
//  - One-time f32->bf16 conversion pass (combined, Wu, Ws, Wp) so the GEMM
//    hot loop does pure b128 data movement + v_wmma_f32_16x16x32_bf16.
//  - Dual GEMM (update+select) shares the A operand.
//  - Double-buffered LDS, one barrier per k-step, global loads for step k+1
//    issued before the WMMAs of step k.
//  - All bf16 operands live in L2 (80 MB << 192 MB), so tile re-reads are
//    L2 hits, keeping the kernels WMMA-bound (~40 us) not HBM-bound.
// ---------------------------------------------------------------------------

typedef __attribute__((ext_vector_type(16))) __bf16        v16bf;
typedef __attribute__((ext_vector_type(8)))  float         v8f;
typedef __attribute__((ext_vector_type(4)))  unsigned int  v4u;

union FragB {
    v4u   q[2];
    v16bf v;
};

#define TILE_M  128
#define TILE_N  128
#define TILE_K  32
#define LDS_PAD 40   // shorts/row = 80 B: rows 16B-aligned, stride 20 dwords
                     // -> conflict-free b128 fragment loads across 16 lanes

__device__ __forceinline__ unsigned int f2bf(float f) {
    unsigned int u = __float_as_uint(f);
    u += 0x7fffu + ((u >> 16) & 1u);       // round-to-nearest-even
    return u >> 16;
}

__device__ __forceinline__ float sigmoidf_(float x) {
    return 1.0f / (1.0f + __expf(-x));
}

// ---------------------------------------------------------------------------
// Elementwise f32 -> bf16 (strided dst so x and h can interleave into
// combined = [x | h] with leading dimension C).
// ---------------------------------------------------------------------------
__global__ __launch_bounds__(256) void convert_f32_to_bf16(
    const float* __restrict__ src, unsigned short* __restrict__ dst,
    long long nquads, int cols4, int dstLd)
{
    long long i = (long long)blockIdx.x * 256 + threadIdx.x;
    if (i >= nquads) return;
    long long r = i / cols4;
    int       c = (int)(i - r * (long long)cols4) * 4;
    const float4 v = *(const float4*)(src + i * 4);
    unsigned int lo = f2bf(v.x) | (f2bf(v.y) << 16);
    unsigned int hi = f2bf(v.z) | (f2bf(v.w) << 16);
    unsigned int* p = (unsigned int*)(dst + r * (long long)dstLd + c);
    p[0] = lo;
    p[1] = hi;
}

// ---------------------------------------------------------------------------
// Tile movement: 128x32 bf16 tile, 256 threads, 32 B (2 x b128) per thread.
// ---------------------------------------------------------------------------
__device__ __forceinline__ void load_tile_regs(
    const unsigned short* __restrict__ src, int ld, int row0, int col0,
    int tid, v4u& ra, v4u& rb)
{
    const int r = tid >> 1;
    const int c = (tid & 1) << 4;
    const unsigned short* p = src + (size_t)(row0 + r) * ld + col0 + c;
    ra = *(const v4u*)p;
    rb = *(const v4u*)(p + 8);
}

__device__ __forceinline__ void store_tile_lds(
    unsigned short* __restrict__ dst, int tid, v4u ra, v4u rb)
{
    const int r = tid >> 1;
    const int c = (tid & 1) << 4;
    unsigned short* p = dst + r * LDS_PAD + c;
    *(v4u*)p       = ra;
    *(v4u*)(p + 8) = rb;
}

// A fragment (16x32 bf16, MxK), ISA layout:
//   lane<16: M=lane, K={0..7,16..23}; lane>=16: M=lane-16, K={8..15,24..31}
//   -> two contiguous 16 B chunks at short-offsets half*8 and 16+half*8.
__device__ __forceinline__ v16bf load_a_frag(const unsigned short* __restrict__ As,
                                             int mBase, int lane)
{
    const int half = lane >> 4;
    const unsigned short* row = As + (mBase + (lane & 15)) * LDS_PAD + (half << 3);
    FragB f;
    f.q[0] = *(const v4u*)row;
    f.q[1] = *(const v4u*)(row + 16);
    return f.v;
}

// B fragment (32x16 bf16, KxN): B[k][n] = W[n][k]; LDS tile holds W rows.
//   lane<16: N=lane, K={0..7,16..23}; lane>=16: N=lane-16, K={8..15,24..31}.
__device__ __forceinline__ v16bf load_b_frag(const unsigned short* __restrict__ Bs,
                                             int nBase, int lane)
{
    const int half = lane >> 4;
    const unsigned short* row = Bs + (nBase + (lane & 15)) * LDS_PAD + (half << 3);
    FragB f;
    f.q[0] = *(const v4u*)row;
    f.q[1] = *(const v4u*)(row + 16);
    return f.v;
}

#define WMMA_BF16(A, Bm, Cacc) \
    __builtin_amdgcn_wmma_f32_16x16x32_bf16(false, (A), false, (Bm), (short)0, (Cacc), false, false)

// ---------------------------------------------------------------------------
// Kernel 1: gates (dual GEMM over combined=[x|h] bf16) + fused epilogue.
//   updB[m,n] = bf16( h[m,n] * sigmoid(accU + bu[n]) )
//   sel [m,n] =       sigmoid(accS + bs[n])
// ---------------------------------------------------------------------------
__global__ __launch_bounds__(256) void gru_gates_kernel(
    const unsigned short* __restrict__ comb,
    const float* __restrict__ h,
    const unsigned short* __restrict__ Wu, const float* __restrict__ bu,
    const unsigned short* __restrict__ Ws, const float* __restrict__ bs,
    unsigned short* __restrict__ updB, float* __restrict__ sel,
    int H, int C)
{
    __shared__ unsigned short smA[2][TILE_M * LDS_PAD];
    __shared__ unsigned short smU[2][TILE_N * LDS_PAD];
    __shared__ unsigned short smS[2][TILE_N * LDS_PAD];

    const int tid   = threadIdx.x;
    const int lane  = tid & 31;
    const int wave  = tid >> 5;
    const int waveM = wave >> 2;   // 0..1 -> 64 rows
    const int waveN = wave & 3;    // 0..3 -> 32 cols
    const int m0    = blockIdx.y * TILE_M;
    const int n0    = blockIdx.x * TILE_N;

    v8f accU[4][2], accS[4][2];
#pragma unroll
    for (int i = 0; i < 4; ++i)
#pragma unroll
        for (int j = 0; j < 2; ++j) {
            accU[i][j] = (v8f){0.f, 0.f, 0.f, 0.f, 0.f, 0.f, 0.f, 0.f};
            accS[i][j] = (v8f){0.f, 0.f, 0.f, 0.f, 0.f, 0.f, 0.f, 0.f};
        }

    const int nk = C / TILE_K;

    v4u rAa, rAb, rUa, rUb, rSa, rSb;
    load_tile_regs(comb, C, m0, 0, tid, rAa, rAb);
    load_tile_regs(Wu,   C, n0, 0, tid, rUa, rUb);
    load_tile_regs(Ws,   C, n0, 0, tid, rSa, rSb);
    store_tile_lds(smA[0], tid, rAa, rAb);
    store_tile_lds(smU[0], tid, rUa, rUb);
    store_tile_lds(smS[0], tid, rSa, rSb);
    __syncthreads();

    for (int kk = 0; kk < nk; ++kk) {
        const int cur = kk & 1;
        const int nxt = cur ^ 1;
        const bool more = (kk + 1) < nk;
        if (more) {
            const int k1 = (kk + 1) * TILE_K;
            load_tile_regs(comb, C, m0, k1, tid, rAa, rAb);
            load_tile_regs(Wu,   C, n0, k1, tid, rUa, rUb);
            load_tile_regs(Ws,   C, n0, k1, tid, rSa, rSb);
        }

        v16bf a[4], bU[2], bS[2];
#pragma unroll
        for (int tm = 0; tm < 4; ++tm)
            a[tm] = load_a_frag(smA[cur], waveM * 64 + tm * 16, lane);
#pragma unroll
        for (int tn = 0; tn < 2; ++tn) {
            bU[tn] = load_b_frag(smU[cur], waveN * 32 + tn * 16, lane);
            bS[tn] = load_b_frag(smS[cur], waveN * 32 + tn * 16, lane);
        }
#pragma unroll
        for (int tm = 0; tm < 4; ++tm)
#pragma unroll
            for (int tn = 0; tn < 2; ++tn) {
                accU[tm][tn] = WMMA_BF16(a[tm], bU[tn], accU[tm][tn]);
                accS[tm][tn] = WMMA_BF16(a[tm], bS[tn], accS[tm][tn]);
            }

        if (more) {
            store_tile_lds(smA[nxt], tid, rAa, rAb);
            store_tile_lds(smU[nxt], tid, rUa, rUb);
            store_tile_lds(smS[nxt], tid, rSa, rSb);
        }
        __syncthreads();
    }

    // Epilogue.  C/D layout: n = lane&15, m = (lane>>4)*8 + vgpr_index.
    const int halfL = lane >> 4;
    const int nl    = lane & 15;
#pragma unroll
    for (int tm = 0; tm < 4; ++tm)
#pragma unroll
        for (int tn = 0; tn < 2; ++tn) {
            const int nG    = n0 + waveN * 32 + tn * 16 + nl;
            const float bun = bu[nG];
            const float bsn = bs[nG];
#pragma unroll
            for (int r = 0; r < 8; ++r) {
                const int mG   = m0 + waveM * 64 + tm * 16 + halfL * 8 + r;
                const size_t o = (size_t)mG * H + nG;
                const float u  = sigmoidf_(accU[tm][tn][r] + bun);
                const float s  = sigmoidf_(accS[tm][tn][r] + bsn);
                updB[o] = (unsigned short)f2bf(h[o] * u);
                sel[o]  = s;
            }
        }
}

// ---------------------------------------------------------------------------
// Kernel 2: predict GEMM over [comb_x_half | updB] + fused blend.
//   p = tanh(acc + bp[n]);  h_new = h + sel*(p - h)
// ---------------------------------------------------------------------------
__global__ __launch_bounds__(256) void gru_predict_kernel(
    const unsigned short* __restrict__ comb,   // x half used (cols < I)
    const unsigned short* __restrict__ updB,   // [B,H] bf16
    const float* __restrict__ h, const float* __restrict__ sel,
    const unsigned short* __restrict__ Wp, const float* __restrict__ bp,
    float* __restrict__ h_new,
    int I, int H, int C)
{
    __shared__ unsigned short smA[2][TILE_M * LDS_PAD];
    __shared__ unsigned short smP[2][TILE_N * LDS_PAD];

    const int tid   = threadIdx.x;
    const int lane  = tid & 31;
    const int wave  = tid >> 5;
    const int waveM = wave >> 2;
    const int waveN = wave & 3;
    const int m0    = blockIdx.y * TILE_M;
    const int n0    = blockIdx.x * TILE_N;

    v8f acc[4][2];
#pragma unroll
    for (int i = 0; i < 4; ++i)
#pragma unroll
        for (int j = 0; j < 2; ++j)
            acc[i][j] = (v8f){0.f, 0.f, 0.f, 0.f, 0.f, 0.f, 0.f, 0.f};

    const int nk = C / TILE_K;

    v4u rAa, rAb, rPa, rPb;
    load_tile_regs(comb, C, m0, 0, tid, rAa, rAb);   // k=0 < I
    load_tile_regs(Wp,   C, n0, 0, tid, rPa, rPb);
    store_tile_lds(smA[0], tid, rAa, rAb);
    store_tile_lds(smP[0], tid, rPa, rPb);
    __syncthreads();

    for (int kk = 0; kk < nk; ++kk) {
        const int cur = kk & 1;
        const int nxt = cur ^ 1;
        const bool more = (kk + 1) < nk;
        if (more) {
            const int k1 = (kk + 1) * TILE_K;
            if (k1 < I) load_tile_regs(comb, C, m0, k1,     tid, rAa, rAb);
            else        load_tile_regs(updB, H, m0, k1 - I, tid, rAa, rAb);
            load_tile_regs(Wp, C, n0, k1, tid, rPa, rPb);
        }

        v16bf a[4], b[2];
#pragma unroll
        for (int tm = 0; tm < 4; ++tm)
            a[tm] = load_a_frag(smA[cur], waveM * 64 + tm * 16, lane);
#pragma unroll
        for (int tn = 0; tn < 2; ++tn)
            b[tn] = load_b_frag(smP[cur], waveN * 32 + tn * 16, lane);
#pragma unroll
        for (int tm = 0; tm < 4; ++tm)
#pragma unroll
            for (int tn = 0; tn < 2; ++tn)
                acc[tm][tn] = WMMA_BF16(a[tm], b[tn], acc[tm][tn]);

        if (more) {
            store_tile_lds(smA[nxt], tid, rAa, rAb);
            store_tile_lds(smP[nxt], tid, rPa, rPb);
        }
        __syncthreads();
    }

    const int halfL = lane >> 4;
    const int nl    = lane & 15;
#pragma unroll
    for (int tm = 0; tm < 4; ++tm)
#pragma unroll
        for (int tn = 0; tn < 2; ++tn) {
            const int nG    = n0 + waveN * 32 + tn * 16 + nl;
            const float bpn = bp[nG];
#pragma unroll
            for (int r = 0; r < 8; ++r) {
                const int mG   = m0 + waveM * 64 + tm * 16 + halfL * 8 + r;
                const size_t o = (size_t)mG * H + nG;
                const float p  = tanhf(acc[tm][tn][r] + bpn);
                const float sv = sel[o];
                const float hv = h[o];
                h_new[o] = hv + sv * (p - hv);
            }
        }
}

// ---------------------------------------------------------------------------
// Host-side launch.
// ---------------------------------------------------------------------------
extern "C" void kernel_launch(void* const* d_in, const int* in_sizes, int n_in,
                              void* d_out, int out_size, void* d_ws, size_t ws_size,
                              hipStream_t stream) {
    const float* x  = (const float*)d_in[0];
    const float* h  = (const float*)d_in[1];
    const float* Wu = (const float*)d_in[2];
    const float* bu = (const float*)d_in[3];
    const float* Ws = (const float*)d_in[4];
    const float* bs = (const float*)d_in[5];
    const float* Wp = (const float*)d_in[6];
    const float* bp = (const float*)d_in[7];
    float* out = (float*)d_out;

    const int H = in_sizes[3];                 // 2048
    const int B = in_sizes[1] / H;             // 4096
    const int C = in_sizes[2] / H;             // 4096
    const int I = in_sizes[0] / B;             // 2048

    // Workspace layout (bf16 operands + f32 select):
    //   comb [B,C] bf16 | WuB [H,C] bf16 | WsB | WpB | updB [B,H] bf16 | sel [B,H] f32
    unsigned short* comb = (unsigned short*)d_ws;
    unsigned short* WuB  = comb + (size_t)B * C;
    unsigned short* WsB  = WuB  + (size_t)H * C;
    unsigned short* WpB  = WsB  + (size_t)H * C;
    unsigned short* updB = WpB  + (size_t)H * C;
    float*          sel  = (float*)(updB + (size_t)B * H);

    dim3 block(256);

    // 1) precision conversion pass (once; GEMMs reread bf16 from L2)
    {
        long long qx = (long long)B * I / 4;
        long long qh = (long long)B * H / 4;
        long long qw = (long long)H * C / 4;
        convert_f32_to_bf16<<<dim3((unsigned)((qx + 255) / 256)), block, 0, stream>>>(
            x, comb, qx, I / 4, C);
        convert_f32_to_bf16<<<dim3((unsigned)((qh + 255) / 256)), block, 0, stream>>>(
            h, comb + I, qh, H / 4, C);
        convert_f32_to_bf16<<<dim3((unsigned)((qw + 255) / 256)), block, 0, stream>>>(
            Wu, WuB, qw, C / 4, C);
        convert_f32_to_bf16<<<dim3((unsigned)((qw + 255) / 256)), block, 0, stream>>>(
            Ws, WsB, qw, C / 4, C);
        convert_f32_to_bf16<<<dim3((unsigned)((qw + 255) / 256)), block, 0, stream>>>(
            Wp, WpB, qw, C / 4, C);
    }

    dim3 grid(H / TILE_N, B / TILE_M);

    // 2) gates: dual GEMM + sigmoid, writes updB (bf16) and sel (f32)
    gru_gates_kernel<<<grid, block, 0, stream>>>(comb, h, WuB, bu, WsB, bs,
                                                 updB, sel, H, C);
    // 3) predict: GEMM over [x | upd] + tanh + blend
    gru_predict_kernel<<<grid, block, 0, stream>>>(comb, updB, h, sel, WpB, bp,
                                                   out, I, H, C);
}
